// DS_HGNN_layer_updata_77721728188419
// MI455X (gfx1250) — matrix operand order repaired and checked by reading, not compiled, not run on hardware
//
#include <hip/hip_runtime.h>

#define BATCH 8
#define NN    1024
#define CC    64
#define MEDGE 1220       // NN + 196 local edges
#define MP    1248       // MEDGE padded to multiple of 32
#define MAXK  64
#define KNN1  11         // K_NEIGS + 1
#define NLOC  196        // (32-5)/2+1 = 14 -> 14*14
#define NLAYERS 8
#define NSTEP 4
#define AV    0.05f
#define EPSV  1e-5f

typedef __attribute__((ext_vector_type(16))) _Float16 v16h;
typedef __attribute__((ext_vector_type(8)))  float    v8f;

// CDNA5 16-bit A-matrix 16x32 K offsets (CDNA5 ISA 7.12.2)
__device__ __forceinline__ int a_koff(int v, int hi) {
  return ((v < 4) ? (2 * v) : (16 + 2 * (v - 4))) + (hi ? 8 : 0);
}
// B-matrix 32x16: same K order as A (VGPR v holds rows {k,k+1} at a_koff(v, hi))
__device__ __forceinline__ int b_koff(int v, int hi) {
  return ((v < 4) ? (2 * v) : (16 + 2 * (v - 4))) + (hi ? 8 : 0);
}

// ---- gfx1250 async copy to LDS (ASYNCcnt-tracked), per-lane 16B ------------
__device__ __forceinline__ void async_b128(unsigned lds_off, const void* g) {
  unsigned long long ga = (unsigned long long)g;
  asm volatile("global_load_async_to_lds_b128 %0, %1, off"
               :: "v"(lds_off), "v"(ga)
               : "memory");
}
__device__ __forceinline__ void wait_async6() {
  asm volatile("s_wait_asynccnt 0x6" ::: "memory");
}
__device__ __forceinline__ void wait_async0() {
  asm volatile("s_wait_asynccnt 0x0" ::: "memory");
}
__device__ __forceinline__ void wait_ds0() {
  asm volatile("s_wait_dscnt 0x0" ::: "memory");
}

// ---------------------------------------------------------------- utilities
__global__ void copy_kernel(const float* __restrict__ src, float* __restrict__ dst, size_t n) {
  size_t i = (size_t)blockIdx.x * blockDim.x + threadIdx.x;
  size_t st = (size_t)gridDim.x * blockDim.x;
  for (; i < n; i += st) dst[i] = src[i];
}
__global__ void zero_i32_kernel(int* __restrict__ p, size_t n) {
  size_t i = (size_t)blockIdx.x * blockDim.x + threadIdx.x;
  size_t st = (size_t)gridDim.x * blockDim.x;
  for (; i < n; i += st) p[i] = 0;
}
__global__ void zero_f32_kernel(float* __restrict__ p, size_t n) {
  size_t i = (size_t)blockIdx.x * blockDim.x + threadIdx.x;
  size_t st = (size_t)gridDim.x * blockDim.x;
  for (; i < n; i += st) p[i] = 0.0f;
}

// ---------------------------------------------------------------- row norms
__global__ void sqnorm_kernel(const float* __restrict__ X, float* __restrict__ sq) {
  int i = blockIdx.x * blockDim.x + threadIdx.x;
  if (i >= BATCH * NN) return;
  const float* x = X + (size_t)i * CC;
  float s = 0.0f;
#pragma unroll
  for (int c = 0; c < CC; ++c) s += x[c] * x[c];
  sq[i] = s;
}

// ---------------------------------------------------------------- D = sq_i + sq_j - 2 * X X^T  (WMMA)
__global__ void dist_wmma_kernel(const float* __restrict__ X, const float* __restrict__ sq,
                                 float* __restrict__ D) {
  const int ti = blockIdx.x * 16, tj = blockIdx.y * 16, b = blockIdx.z;
  const int L = threadIdx.x, m = L & 15, hi = L >> 4;
  const float* Xb = X + (size_t)b * NN * CC;
  __builtin_prefetch(&Xb[(ti + m) * CC], 0, 1);
  v8f acc = {};
#pragma unroll
  for (int k0 = 0; k0 < CC; k0 += 32) {
    v16h af, bf;
#pragma unroll
    for (int v = 0; v < 8; ++v) {
      int ka = a_koff(v, hi);
      af[2 * v]     = (_Float16)Xb[(ti + m) * CC + k0 + ka];
      af[2 * v + 1] = (_Float16)Xb[(ti + m) * CC + k0 + ka + 1];
      int kb = b_koff(v, hi);
      bf[2 * v]     = (_Float16)Xb[(tj + m) * CC + k0 + kb];
      bf[2 * v + 1] = (_Float16)Xb[(tj + m) * CC + k0 + kb + 1];
    }
    acc = __builtin_amdgcn_wmma_f32_16x16x32_f16(false, af, false, bf, (short)0, acc, false, false);
  }
  const float* sqb = sq + b * NN;
  float* Db = D + (size_t)b * NN * NN;
#pragma unroll
  for (int v = 0; v < 8; ++v) {
    int r = ti + v + hi * 8;
    int c = tj + m;
    Db[(size_t)r * NN + c] = sqb[r] + sqb[c] - 2.0f * acc[v];
  }
}

// ---------------------------------------------------------------- top-64 smallest per row (1 wave32 / row)
__global__ void topk_kernel(const float* __restrict__ Dist, int* __restrict__ Inds) {
  const int row = blockIdx.x, b = blockIdx.y, L = threadIdx.x;
  const float* d = Dist + ((size_t)b * NN + row) * NN;
  float v[32];
#pragma unroll
  for (int t = 0; t < 32; ++t) v[t] = d[t * 32 + L];
  int* out = Inds + ((size_t)b * NN + row) * MAXK;
  for (int it = 0; it < MAXK; ++it) {
    float bv = v[0];
    int bt = 0;
#pragma unroll
    for (int t = 1; t < 32; ++t) {
      if (v[t] < bv) { bv = v[t]; bt = t; }
    }
    int bj = bt * 32 + L;
#pragma unroll
    for (int off = 16; off >= 1; off >>= 1) {
      float ov = __shfl_down(bv, off, 32);
      int   oj = __shfl_down(bj, off, 32);
      if (ov < bv || (ov == bv && oj < bj)) { bv = ov; bj = oj; }
    }
    int j0 = __shfl(bj, 0, 32);
    if (L == 0) out[it] = j0;
    if ((j0 & 31) == L) {
      int slot = j0 >> 5;
#pragma unroll
      for (int t = 0; t < 32; ++t)
        if (t == slot) v[t] = 3.4e38f;
    }
  }
}

// ---------------------------------------------------------------- in-degree of (K+1)-NN graph
__global__ void degree_kernel(const int* __restrict__ Inds, int* __restrict__ DvC) {
  int i = blockIdx.x * blockDim.x + threadIdx.x;
  if (i >= BATCH * NN) return;
  const int* row = Inds + (size_t)i * MAXK;
  int b = i / NN;
#pragma unroll
  for (int k = 0; k < KNN1; ++k) atomicAdd(&DvC[b * NN + row[k]], 1);
}

// ---------------------------------------------------------------- scatter knn edges into dense H (f16)
__global__ void buildH_knn_kernel(const int* __restrict__ Inds, const int* __restrict__ DvC,
                                  _Float16* __restrict__ Hh, int* __restrict__ DVc,
                                  float* __restrict__ DEi) {
  int i = blockIdx.x * blockDim.x + threadIdx.x;
  if (i >= BATCH * NN) return;
  int b = i / NN, e = i % NN;
  int cnt = DvC[i];
  cnt = cnt < 1 ? 1 : (cnt > MAXK ? MAXK : cnt);
  DEi[(size_t)b * MP + e] = 1.0f / (float)cnt;
  const int* row = Inds + (size_t)i * MAXK;
  _Float16* Hb = Hh + (size_t)b * NN * MP;
  for (int k = 0; k < cnt; ++k) {
    int j = row[k];
    Hb[(size_t)j * MP + e] = (_Float16)1.0f;
    atomicAdd(&DVc[b * NN + j], 1);
  }
}

// ---------------------------------------------------------------- local 5x5/stride2 patch edges
__global__ void buildH_local_kernel(_Float16* __restrict__ Hh, int* __restrict__ DVc,
                                    float* __restrict__ DEi) {
  int i = blockIdx.x * blockDim.x + threadIdx.x;
  if (i >= BATCH * NLOC) return;
  int b = i / NLOC, le = i % NLOC;
  int r = le / 14, c = le % 14;
  DEi[(size_t)b * MP + NN + le] = 1.0f / 25.0f;
  _Float16* Hb = Hh + (size_t)b * NN * MP;
#pragma unroll
  for (int dr = 0; dr < 5; ++dr)
#pragma unroll
    for (int dc = 0; dc < 5; ++dc) {
      int j = (2 * r + dr) * 32 + (2 * c + dc);
      Hb[(size_t)j * MP + NN + le] = (_Float16)1.0f;
      atomicAdd(&DVc[b * NN + j], 1);
    }
}

__global__ void dvinv_kernel(const int* __restrict__ DVc, float* __restrict__ DVi) {
  int i = blockIdx.x * blockDim.x + threadIdx.x;
  if (i >= BATCH * NN) return;
  int d = DVc[i];
  DVi[i] = 1.0f / (float)(d < 1 ? 1 : d);
}

// ---------------------------------------------------------------- E = diag(1/DE) * H^T * X
// Async-to-LDS double-buffered WMMA GEMM. Tile: 16 edges x 16 chans, K=32/step.
__global__ void egemm_kernel(const _Float16* __restrict__ Hh, const float* __restrict__ X,
                             const float* __restrict__ DEi, float* __restrict__ E) {
  const int e0 = blockIdx.x * 16, c0 = blockIdx.y * 16, b = blockIdx.z;
  const int L = threadIdx.x, m = L & 15, hi = L >> 4;
  const _Float16* Hb = Hh + (size_t)b * NN * MP;
  const float* Xb = X + (size_t)b * NN * CC;

  __shared__ _Float16 shA[2][32][16];   // H[k][e0..e0+15]
  __shared__ float    shB[2][32][16];   // X[k][c0..c0+15]
  unsigned ldsA[2] = { (unsigned)(size_t)&shA[0][0][0], (unsigned)(size_t)&shA[1][0][0] };
  unsigned ldsB[2] = { (unsigned)(size_t)&shB[0][0][0], (unsigned)(size_t)&shB[1][0][0] };

  // stage(kt -> buffer s): 6 async b128 copies (2 for A, 4 for B)
  auto stage = [&](int s, int kt) {
    wait_ds0();  // close LDS-read vs async-LDS-write hazard on buffer reuse
#pragma unroll
    for (int q = 0; q < 2; ++q)
      async_b128(ldsA[s] + (unsigned)(L * 32 + q * 16),
                 (const void*)(Hb + (size_t)(kt + L) * MP + e0 + q * 8));
#pragma unroll
    for (int q = 0; q < 4; ++q)
      async_b128(ldsB[s] + (unsigned)(L * 64 + q * 16),
                 (const void*)(Xb + (size_t)(kt + L) * CC + c0 + q * 4));
  };

  stage(0, 0);
  v8f acc = {};
#pragma unroll 2
  for (int k0 = 0; k0 < NN; k0 += 32) {
    int buf = (k0 >> 5) & 1;
    if (k0 + 32 < NN) { stage(buf ^ 1, k0 + 32); wait_async6(); }
    else              { wait_async0(); }
    v16h af, bf;
#pragma unroll
    for (int v = 0; v < 8; ++v) {
      int ka = a_koff(v, hi);
      af[2 * v]     = shA[buf][ka][m];
      af[2 * v + 1] = shA[buf][ka + 1][m];
      int kb = b_koff(v, hi);
      bf[2 * v]     = (_Float16)shB[buf][kb][m];
      bf[2 * v + 1] = (_Float16)shB[buf][kb + 1][m];
    }
    acc = __builtin_amdgcn_wmma_f32_16x16x32_f16(false, af, false, bf, (short)0, acc, false, false);
  }
  const float* dei = DEi + (size_t)b * MP;
  float* Eb = E + (size_t)b * MP * CC;
#pragma unroll
  for (int v = 0; v < 8; ++v) {
    int e = e0 + v + hi * 8;
    Eb[(size_t)e * CC + c0 + m] = acc[v] * dei[e];
  }
}

// ---------------------------------------------------------------- Y = S + relu(S @ W^T + bias)  (WMMA)
__global__ void theta_kernel(const float* __restrict__ S, const float* __restrict__ W,
                             const float* __restrict__ bias, float* __restrict__ Y) {
  const int r0 = blockIdx.x * 16, c0 = blockIdx.y * 16;
  const int L = threadIdx.x, m = L & 15, hi = L >> 4;
  v8f acc = {};
#pragma unroll
  for (int k0 = 0; k0 < CC; k0 += 32) {
    v16h af, bf;
#pragma unroll
    for (int v = 0; v < 8; ++v) {
      int ka = a_koff(v, hi);
      af[2 * v]     = (_Float16)S[(size_t)(r0 + m) * CC + k0 + ka];
      af[2 * v + 1] = (_Float16)S[(size_t)(r0 + m) * CC + k0 + ka + 1];
      int kb = b_koff(v, hi);
      bf[2 * v]     = (_Float16)W[(size_t)(c0 + m) * CC + k0 + kb];   // B[k][n] = W[n][k]
      bf[2 * v + 1] = (_Float16)W[(size_t)(c0 + m) * CC + k0 + kb + 1];
    }
    acc = __builtin_amdgcn_wmma_f32_16x16x32_f16(false, af, false, bf, (short)0, acc, false, false);
  }
  float bb = bias[c0 + m];
#pragma unroll
  for (int v = 0; v < 8; ++v) {
    size_t idx = (size_t)(r0 + v + hi * 8) * CC + c0 + m;
    Y[idx] = S[idx] + fmaxf(acc[v] + bb, 0.0f);
  }
}

// ---------------------------------------------------------------- per-index BN over (batch, channel)
__global__ void bn_kernel(const float* __restrict__ Y, float* __restrict__ X, int rows) {
  const int n = blockIdx.x, tid = threadIdx.x;
  __shared__ float ssum[256], ssq[256];
  float s = 0.0f, q = 0.0f;
  for (int i = tid; i < BATCH * CC; i += 256) {
    int b = i / CC, c = i % CC;
    float x = Y[((size_t)b * rows + n) * CC + c];
    s += x; q += x * x;
  }
  ssum[tid] = s; ssq[tid] = q;
  __syncthreads();
  for (int off = 128; off; off >>= 1) {
    if (tid < off) { ssum[tid] += ssum[tid + off]; ssq[tid] += ssq[tid + off]; }
    __syncthreads();
  }
  float mean = ssum[0] * (1.0f / (BATCH * CC));
  float var  = ssq[0] * (1.0f / (BATCH * CC)) - mean * mean;
  float inv  = rsqrtf(var + EPSV);
  for (int i = tid; i < BATCH * CC; i += 256) {
    int b = i / CC, c = i % CC;
    size_t idx = ((size_t)b * rows + n) * CC + c;
    X[idx] = (Y[idx] - mean) * inv;
  }
}

// ---------------------------------------------------------------- X = (1-a)X + a * diag(1/DV) * H * E
// Async-to-LDS double-buffered WMMA GEMM over K = MP edges.
__global__ void xupdate_kernel(const _Float16* __restrict__ Hh, const float* __restrict__ E,
                               const float* __restrict__ DVi, float* __restrict__ X) {
  const int j0 = blockIdx.x * 16, c0 = blockIdx.y * 16, b = blockIdx.z;
  const int L = threadIdx.x, m = L & 15, hi = L >> 4;
  const _Float16* Hb = Hh + (size_t)b * NN * MP;
  const float* Eb = E + (size_t)b * MP * CC;

  __shared__ _Float16 shA[2][16][32];   // H[j0+r][k..k+31]
  __shared__ float    shB[2][32][16];   // E[k][c0..c0+15]
  unsigned ldsA[2] = { (unsigned)(size_t)&shA[0][0][0], (unsigned)(size_t)&shA[1][0][0] };
  unsigned ldsB[2] = { (unsigned)(size_t)&shB[0][0][0], (unsigned)(size_t)&shB[1][0][0] };

  auto stage = [&](int s, int kt) {
    wait_ds0();
    int row = L >> 1;
#pragma unroll
    for (int q = 0; q < 2; ++q) {
      int ch = (L & 1) + 2 * q;   // 4 x 16B chunks per 64B row
      async_b128(ldsA[s] + (unsigned)(row * 64 + ch * 16),
                 (const void*)(Hb + (size_t)(j0 + row) * MP + kt + ch * 8));
    }
#pragma unroll
    for (int q = 0; q < 4; ++q)
      async_b128(ldsB[s] + (unsigned)(L * 64 + q * 16),
                 (const void*)(Eb + (size_t)(kt + L) * CC + c0 + q * 4));
  };

  stage(0, 0);
  v8f acc = {};
#pragma unroll 2
  for (int k0 = 0; k0 < MP; k0 += 32) {
    int buf = (k0 >> 5) & 1;
    if (k0 + 32 < MP) { stage(buf ^ 1, k0 + 32); wait_async6(); }
    else              { wait_async0(); }
    v16h af, bf;
#pragma unroll
    for (int v = 0; v < 8; ++v) {
      int ka = a_koff(v, hi);
      af[2 * v]     = shA[buf][m][ka];
      af[2 * v + 1] = shA[buf][m][ka + 1];
      int kb = b_koff(v, hi);
      bf[2 * v]     = (_Float16)shB[buf][kb][m];
      bf[2 * v + 1] = (_Float16)shB[buf][kb + 1][m];
    }
    acc = __builtin_amdgcn_wmma_f32_16x16x32_f16(false, af, false, bf, (short)0, acc, false, false);
  }
  float* Xb = X + (size_t)b * NN * CC;
  const float* dvi = DVi + b * NN;
#pragma unroll
  for (int v = 0; v < 8; ++v) {
    int j = j0 + v + hi * 8;
    size_t idx = (size_t)j * CC + c0 + m;
    Xb[idx] = (1.0f - AV) * Xb[idx] + AV * acc[v] * dvi[j];
  }
}

// ---------------------------------------------------------------- head: per-channel stats then relu + residual
__global__ void headstats_kernel(const float* __restrict__ X, float* __restrict__ mv) {
  const int c = blockIdx.x, tid = threadIdx.x;
  __shared__ float ssum[256], ssq[256];
  float s = 0.0f, q = 0.0f;
  for (int i = tid; i < BATCH * NN; i += 256) {
    float x = X[(size_t)i * CC + c];
    s += x; q += x * x;
  }
  ssum[tid] = s; ssq[tid] = q;
  __syncthreads();
  for (int off = 128; off; off >>= 1) {
    if (tid < off) { ssum[tid] += ssum[tid + off]; ssq[tid] += ssq[tid + off]; }
    __syncthreads();
  }
  if (tid == 0) {
    float mean = ssum[0] * (1.0f / (BATCH * NN));
    float var  = ssq[0] * (1.0f / (BATCH * NN)) - mean * mean;
    mv[c] = mean;
    mv[CC + c] = var;
  }
}

__global__ void headout_kernel(const float* __restrict__ X, const float* __restrict__ Xin,
                               const float* __restrict__ mv, const float* __restrict__ g,
                               const float* __restrict__ be, float* __restrict__ out) {
  int i = blockIdx.x * blockDim.x + threadIdx.x;
  if (i >= BATCH * NN * CC) return;
  int c = i % CC;
  float inv = rsqrtf(mv[CC + c] + EPSV);
  float xn = (X[i] - mv[c]) * inv * g[c] + be[c];
  out[i] = fmaxf(xn, 0.0f) + Xin[i];
}

// ---------------------------------------------------------------- host orchestration
extern "C" void kernel_launch(void* const* d_in, const int* in_sizes, int n_in,
                              void* d_out, int out_size, void* d_ws, size_t ws_size,
                              hipStream_t stream) {
  const float* Xin = (const float*)d_in[0];
  const float* tvw = (const float*)d_in[1];
  const float* tvb = (const float*)d_in[2];
  const float* tew = (const float*)d_in[3];
  const float* teb = (const float*)d_in[4];
  const float* gam = (const float*)d_in[5];
  const float* bet = (const float*)d_in[6];
  // d_in[7] local_H is regenerated analytically on device
  (void)in_sizes; (void)n_in; (void)out_size; (void)ws_size;

  char* ws = (char*)d_ws;
  size_t off = 0;
  auto salloc = [&](size_t bytes) -> char* {
    char* p = ws + off;
    off += (bytes + 255) & ~((size_t)255);
    return p;
  };
  float*    Xcur = (float*)salloc((size_t)BATCH * NN * CC * 4);
  float*    Dist = (float*)salloc((size_t)BATCH * NN * NN * 4);
  float*    Ebuf = (float*)salloc((size_t)BATCH * MP * CC * 4);
  float*    Ybuf = (float*)salloc((size_t)BATCH * MP * CC * 4);
  _Float16* Hh   = (_Float16*)salloc((size_t)BATCH * NN * MP * 2);
  int*      Inds = (int*)salloc((size_t)BATCH * NN * MAXK * 4);
  int*      DvC  = (int*)salloc((size_t)BATCH * NN * 4);
  int*      DVc  = (int*)salloc((size_t)BATCH * NN * 4);
  float*    DEi  = (float*)salloc((size_t)BATCH * MP * 4);
  float*    DVi  = (float*)salloc((size_t)BATCH * NN * 4);
  float*    sqn  = (float*)salloc((size_t)BATCH * NN * 4);
  float*    mv   = (float*)salloc((size_t)2 * CC * 4);

  copy_kernel<<<2048, 256, 0, stream>>>(Xin, Xcur, (size_t)BATCH * NN * CC);

  for (int layer = 0; layer < NLAYERS; ++layer) {
    // clear per-layer scatter targets
    zero_i32_kernel<<<4096, 256, 0, stream>>>((int*)Hh, (size_t)BATCH * NN * MP / 2);
    zero_i32_kernel<<<32, 256, 0, stream>>>(DvC, (size_t)BATCH * NN);
    zero_i32_kernel<<<32, 256, 0, stream>>>(DVc, (size_t)BATCH * NN);
    zero_f32_kernel<<<40, 256, 0, stream>>>(DEi, (size_t)BATCH * MP);

    // hypergraph generation
    sqnorm_kernel<<<32, 256, 0, stream>>>(Xcur, sqn);
    dist_wmma_kernel<<<dim3(NN / 16, NN / 16, BATCH), 32, 0, stream>>>(Xcur, sqn, Dist);
    topk_kernel<<<dim3(NN, BATCH), 32, 0, stream>>>(Dist, Inds);
    degree_kernel<<<32, 256, 0, stream>>>(Inds, DvC);
    buildH_knn_kernel<<<32, 256, 0, stream>>>(Inds, DvC, Hh, DVc, DEi);
    buildH_local_kernel<<<7, 256, 0, stream>>>(Hh, DVc, DEi);
    dvinv_kernel<<<32, 256, 0, stream>>>(DVc, DVi);

    // hyperedge features: E = diag(1/DE) H^T X
    egemm_kernel<<<dim3(MP / 16, CC / 16, BATCH), 32, 0, stream>>>(Hh, Xcur, DEi, Ebuf);

    if (layer % NSTEP == 0) {
      theta_kernel<<<dim3(BATCH * NN / 16, CC / 16), 32, 0, stream>>>(Xcur, tvw, tvb, Ybuf);
      bn_kernel<<<NN, 256, 0, stream>>>(Ybuf, Xcur, NN);
      theta_kernel<<<dim3(BATCH * MP / 16, CC / 16), 32, 0, stream>>>(Ebuf, tew, teb, Ybuf);
      bn_kernel<<<MP, 256, 0, stream>>>(Ybuf, Ebuf, MP);
    }

    // vertex smoothing: X = (1-a)X + a diag(1/DV) H E
    xupdate_kernel<<<dim3(NN / 16, CC / 16, BATCH), 32, 0, stream>>>(Hh, Ebuf, DVi, Xcur);
  }

  headstats_kernel<<<CC, 256, 0, stream>>>(Xcur, mv);
  headout_kernel<<<2048, 256, 0, stream>>>(Xcur, Xin, mv, gam, bet, (float*)d_out);
}
